// CompositionalAttention_22995254902805
// MI455X (gfx1250) — compile-verified, matrix operand order checked by reading, not run on hardware
//
#include <hip/hip_runtime.h>
#include <hip/hip_bf16.h>

// ---------------------------------------------------------------------------
// Compositional attention on MI455X (gfx1250), bf16 WMMA everywhere,
// async global->LDS staging (ASYNCcnt path) for all matrix tiles.
// S=8 search heads, R=2 retrievals, DH=64, b=2, n=2048, dim=512.
// ---------------------------------------------------------------------------

#define B_SZ   2
#define N_SEQ  2048
#define DMODEL 512
#define S_H    8
#define R_H    2
#define DIMH   64

typedef __attribute__((ext_vector_type(16))) __bf16       v16bf;
typedef __attribute__((ext_vector_type(8)))  float        v8f;
typedef __attribute__((ext_vector_type(4)))  unsigned int u32x4;

union Frag {
    v16bf v;
    u32x4 q[2];
};

__device__ inline v8f wmma_bf16(const Frag& a, const Frag& b, v8f c) {
    return __builtin_amdgcn_wmma_f32_16x16x32_bf16(
        /*neg_a=*/false, a.v, /*neg_b=*/false, b.v,
        /*c_mod=*/(short)0, c, /*reuse_a=*/false, /*reuse_b=*/false);
}

// Async DMA of 16 bytes global -> LDS (no VGPR data path, ASYNCcnt-tracked).
// LDS destination byte offset = low 32 bits of the generic shared pointer.
__device__ inline void async_load_b128(void* lds_dst, const void* gsrc) {
    unsigned loff = (unsigned)(size_t)lds_dst;
    asm volatile("global_load_async_to_lds_b128 %0, %1, off"
                 :: "v"(loff), "v"(gsrc)
                 : "memory");
}

__device__ inline void wait_async() {
    asm volatile("s_wait_asynccnt 0x0" ::: "memory");
}

// ---------------------------------------------------------------------------
// Elementwise f32 -> bf16 convert
// ---------------------------------------------------------------------------
__global__ void cvt_f32_bf16_kernel(const float* __restrict__ src,
                                    __bf16* __restrict__ dst, int count) {
    int idx = blockIdx.x * 256 + threadIdx.x;
    if (idx < count) dst[idx] = (__bf16)src[idx];
}

// W (K x N) f32  ->  Wt (N x K) bf16
__global__ void transpose_cvt_kernel(const float* __restrict__ W,
                                     __bf16* __restrict__ Wt, int K, int N) {
    int idx = blockIdx.x * 256 + threadIdx.x;
    if (idx < N * K) {
        int n = idx / K;
        int k = idx - n * K;
        Wt[idx] = (__bf16)W[(size_t)k * N + n];
    }
}

// rv (b, n, R*DH) bf16  ->  rv_t (b, R, DH, n) bf16
__global__ void rv_transpose_kernel(const __bf16* __restrict__ rv,
                                    __bf16* __restrict__ rvt) {
    int idx = blockIdx.x * 256 + threadIdx.x;
    const int total = B_SZ * R_H * DIMH * N_SEQ;
    if (idx < total) {
        int j   = idx % N_SEQ;
        int rem = idx / N_SEQ;
        int d   = rem % DIMH;  rem /= DIMH;
        int r   = rem % R_H;
        int b   = rem / R_H;
        rvt[idx] = rv[(size_t)(b * N_SEQ + j) * (R_H * DIMH) + r * DIMH + d];
    }
}

// ---------------------------------------------------------------------------
// Generic WMMA GEMM: C(MxN) = A(MxK) @ B(KxN), with B supplied transposed
// as Bt(NxK).  A/Bt bf16 row-major.  Optional f32 and/or bf16 output, scaled.
// Block = 128 threads (4 waves), 64x64 tile, K-step 32.
// M, N, K must be multiples of 64/64/32 (true for all uses here).
// ---------------------------------------------------------------------------
__global__ __launch_bounds__(128)
void gemm_bf16_kernel(const __bf16* __restrict__ A,
                      const __bf16* __restrict__ Bt,
                      float* __restrict__ Cf,
                      __bf16* __restrict__ Ch,
                      int M, int N, int K, float scale) {
    const int m0   = blockIdx.x * 64;
    const int n0   = blockIdx.y * 64;
    const int t    = threadIdx.x;
    const int wave = t >> 5;
    const int lane = t & 31;
    const int col  = lane & 15;
    const int half = lane >> 4;      // which 16-lane half
    const int rbase = half * 8;      // C-layout row base
    const int koff  = half * 8;      // A/B fragment K sub-offset

    __shared__ __align__(16) __bf16 lA[64 * 40];
    __shared__ __align__(16) __bf16 lB[64 * 40];

    v8f acc[4];
#pragma unroll
    for (int nt = 0; nt < 4; ++nt) acc[nt] = (v8f)0.f;

    for (int kk = 0; kk < K; kk += 32) {
        __syncthreads();
#pragma unroll
        for (int it = 0; it < 2; ++it) {
            int idx = t + it * 128;          // 0..255
            int row = idx >> 2;              // 0..63
            int cc  = idx & 3;               // 8-elem chunk within 32 cols
            async_load_b128(&lA[row * 40 + cc * 8],
                            A + (size_t)(m0 + row) * K + kk + cc * 8);
            async_load_b128(&lB[row * 40 + cc * 8],
                            Bt + (size_t)(n0 + row) * K + kk + cc * 8);
        }
        wait_async();
        __syncthreads();

        Frag af;
        {
            const __bf16* ap = &lA[(wave * 16 + col) * 40 + koff];
            af.q[0] = *(const u32x4*)ap;
            af.q[1] = *(const u32x4*)(ap + 16);
        }
#pragma unroll
        for (int nt = 0; nt < 4; ++nt) {
            Frag bf_;
            const __bf16* bp = &lB[(nt * 16 + col) * 40 + koff];
            bf_.q[0] = *(const u32x4*)bp;
            bf_.q[1] = *(const u32x4*)(bp + 16);
            acc[nt] = wmma_bf16(af, bf_, acc[nt]);
        }
    }

#pragma unroll
    for (int nt = 0; nt < 4; ++nt) {
#pragma unroll
        for (int e = 0; e < 8; ++e) {
            int mrow = m0 + wave * 16 + rbase + e;
            int ncol = n0 + nt * 16 + col;
            float vv = acc[nt][e] * scale;
            if (Cf) Cf[(size_t)mrow * N + ncol] = vv;
            if (Ch) Ch[(size_t)mrow * N + ncol] = (__bf16)vv;
        }
    }
}

// ---------------------------------------------------------------------------
// Fused flash-style attention + retrieval composition.
//   grid.x = n/64 (row tile), grid.y = b*S.  128 threads = 4 waves,
//   each wave owns a 16-row strip of the 64-row block.
// Inputs:
//   qh (b,n,512) bf16, pre-scaled by DH^-0.5 (head s at cols s*64..)
//   kh (b,n,512) bf16
//   vt (b,R,DH,n) bf16 (rv transposed so V B-fragments load contiguous j)
//   rqf (b,n,512) f32, pre-scaled
//   wkret (64,64) f32
// Output: outsd (b,n,512) bf16  — the "bsid" tensor, head-major cols.
// ---------------------------------------------------------------------------
__global__ __launch_bounds__(128)
void attn_kernel(const __bf16* __restrict__ qh,
                 const __bf16* __restrict__ kh,
                 const __bf16* __restrict__ vt,
                 const float*  __restrict__ rqf,
                 const float*  __restrict__ wkret,
                 __bf16* __restrict__ outsd) {
    const int i0   = blockIdx.x * 64;
    const int bs   = blockIdx.y;
    const int b    = bs / S_H;
    const int s    = bs - b * S_H;
    const int t    = threadIdx.x;
    const int wave = t >> 5;
    const int lane = t & 31;
    const int col  = lane & 15;
    const int half = lane >> 4;
    const int rbase = half * 8;
    const int koff  = half * 8;

    __shared__ __align__(16) __bf16 ldsK[64 * 72];          // K tile (j x d)
    __shared__ __align__(16) __bf16 ldsV[2][64 * 72];       // V tiles (d x j)
    __shared__ __align__(16) __bf16 ldsP[4][16 * 72];       // per-wave P strip
    __shared__ __align__(16) float  ldsR[4][2][16 * 64];    // retrieved strips
    __shared__ __align__(16) float  ldsQ[64 * 64];          // rq rows (f32)

    // Stage rq rows for this block into LDS (used only by the epilogue).
#pragma unroll
    for (int it = 0; it < 32; ++it) {
        int idx = t + it * 128;                  // 0..4095
        int row = idx >> 6;
        int d   = idx & 63;
        ldsQ[idx] = rqf[(size_t)(b * N_SEQ + i0 + row) * DMODEL + s * DIMH + d];
    }

    // Q fragments for this wave's 16-row strip (held in registers all loop).
    Frag qf[2];
    {
        const __bf16* qrow =
            qh + (size_t)(b * N_SEQ + i0 + wave * 16 + col) * DMODEL + s * DIMH;
#pragma unroll
        for (int c = 0; c < 2; ++c) {
            qf[c].q[0] = *(const u32x4*)(qrow + c * 32 + koff);
            qf[c].q[1] = *(const u32x4*)(qrow + c * 32 + koff + 16);
        }
    }

    v8f accR[2][4];
#pragma unroll
    for (int r = 0; r < 2; ++r)
#pragma unroll
        for (int dt = 0; dt < 4; ++dt) accR[r][dt] = (v8f)0.f;

    float m[8], l[8];
#pragma unroll
    for (int e = 0; e < 8; ++e) { m[e] = -3.0e38f; l[e] = 0.f; }

    for (int j0 = 0; j0 < N_SEQ; j0 += 64) {
        __syncthreads();
        // Stage K tile (64j x 64d) and both V tiles (64d x 64j) into LDS
        // via the async global->LDS DMA path (ASYNCcnt).
#pragma unroll
        for (int it = 0; it < 4; ++it) {
            int idx = t + it * 128;          // 0..511
            int row = idx >> 3;              // 0..63
            int cc  = idx & 7;               // 8-elem chunk of 64 cols
            async_load_b128(&ldsK[row * 72 + cc * 8],
                            kh + (size_t)(b * N_SEQ + j0 + row) * DMODEL +
                                s * DIMH + cc * 8);
            async_load_b128(&ldsV[0][row * 72 + cc * 8],
                            vt + ((size_t)(b * R_H + 0) * DIMH + row) * N_SEQ +
                                j0 + cc * 8);
            async_load_b128(&ldsV[1][row * 72 + cc * 8],
                            vt + ((size_t)(b * R_H + 1) * DIMH + row) * N_SEQ +
                                j0 + cc * 8);
        }
        wait_async();
        __syncthreads();

        // S strip = Q @ K^T  (16 rows x 64 j) via WMMA.
        v8f sacc[4];
#pragma unroll
        for (int jt = 0; jt < 4; ++jt) sacc[jt] = (v8f)0.f;
#pragma unroll
        for (int jt = 0; jt < 4; ++jt) {
            const __bf16* kb = &ldsK[(jt * 16 + col) * 72 + koff];
#pragma unroll
            for (int c = 0; c < 2; ++c) {
                Frag kf;
                kf.q[0] = *(const u32x4*)(kb + c * 32);
                kf.q[1] = *(const u32x4*)(kb + c * 32 + 16);
                sacc[jt] = wmma_bf16(qf[c], kf, sacc[jt]);
            }
        }

        // Online softmax (per row of the strip).
        float pr[4][8];
#pragma unroll
        for (int e = 0; e < 8; ++e) {
            float mx = sacc[0][e];
#pragma unroll
            for (int jt = 1; jt < 4; ++jt) mx = fmaxf(mx, sacc[jt][e]);
#pragma unroll
            for (int off = 1; off < 16; off <<= 1)
                mx = fmaxf(mx, __shfl_xor(mx, off));
            float mnew  = fmaxf(m[e], mx);
            float alpha = __expf(m[e] - mnew);
            m[e] = mnew;
            float rsum = 0.f;
#pragma unroll
            for (int jt = 0; jt < 4; ++jt) {
                float p = __expf(sacc[jt][e] - mnew);
                pr[jt][e] = p;
                rsum += p;
            }
#pragma unroll
            for (int off = 1; off < 16; off <<= 1)
                rsum += __shfl_xor(rsum, off);
            l[e] = l[e] * alpha + rsum;
#pragma unroll
            for (int r = 0; r < 2; ++r)
#pragma unroll
                for (int dt = 0; dt < 4; ++dt) accR[r][dt][e] *= alpha;
        }

        // Spill P strip to this wave's LDS region (bf16), reload as A-frags.
        __bf16* pw = &ldsP[wave][0];
#pragma unroll
        for (int jt = 0; jt < 4; ++jt)
#pragma unroll
            for (int e = 0; e < 8; ++e)
                pw[(rbase + e) * 72 + jt * 16 + col] = (__bf16)pr[jt][e];

        Frag pf[2];
        {
            const __bf16* pb = &ldsP[wave][col * 72 + koff];
#pragma unroll
            for (int c = 0; c < 2; ++c) {
                pf[c].q[0] = *(const u32x4*)(pb + c * 32);
                pf[c].q[1] = *(const u32x4*)(pb + c * 32 + 16);
            }
        }

        // accR[r] += P @ V_r   via WMMA.
#pragma unroll
        for (int r = 0; r < 2; ++r) {
#pragma unroll
            for (int dt = 0; dt < 4; ++dt) {
                const __bf16* vb = &ldsV[r][(dt * 16 + col) * 72 + koff];
#pragma unroll
                for (int c = 0; c < 2; ++c) {
                    Frag vf;
                    vf.q[0] = *(const u32x4*)(vb + c * 32);
                    vf.q[1] = *(const u32x4*)(vb + c * 32 + 16);
                    accR[r][dt] = wmma_bf16(pf[c], vf, accR[r][dt]);
                }
            }
        }
    }

    // Normalize and park retrieved strips in LDS for the epilogue.
#pragma unroll
    for (int e = 0; e < 8; ++e) {
        float inv = 1.f / l[e];
#pragma unroll
        for (int r = 0; r < 2; ++r)
#pragma unroll
            for (int dt = 0; dt < 4; ++dt)
                ldsR[wave][r][(rbase + e) * 64 + dt * 16 + col] =
                    accR[r][dt][e] * inv;
    }
    __syncthreads();

    // Retrieval composition epilogue: one thread per row.
    //   sim_r = rq . (retrieved_r @ wk_ret) = retrieved_r . (wk_ret @ rq)
    if (t < 64) {
        const int w  = t >> 4;
        const int rr = t & 15;
        const float* r0 = &ldsR[w][0][rr * 64];
        const float* r1 = &ldsR[w][1][rr * 64];
        const float* qv = &ldsQ[t * 64];

        float sim0 = 0.f, sim1 = 0.f;
        for (int d = 0; d < 64; ++d) {
            const float* wrow = wkret + d * 64;   // uniform: scalar loads
            float td = 0.f;
#pragma unroll 8
            for (int dp = 0; dp < 64; ++dp) td += wrow[dp] * qv[dp];
            sim0 += r0[d] * td;
            sim1 += r1[d] * td;
        }
        float mm = fmaxf(sim0, sim1);
        float e0 = __expf(sim0 - mm), e1 = __expf(sim1 - mm);
        float w0 = e0 / (e0 + e1), w1 = 1.f - w0;

        __bf16* op = outsd + (size_t)(b * N_SEQ + i0 + t) * DMODEL + s * DIMH;
#pragma unroll
        for (int d = 0; d < 64; ++d)
            op[d] = (__bf16)(w0 * r0[d] + w1 * r1[d]);
    }
}

// ---------------------------------------------------------------------------
extern "C" void kernel_launch(void* const* d_in, const int* in_sizes, int n_in,
                              void* d_out, int out_size, void* d_ws, size_t ws_size,
                              hipStream_t stream) {
    (void)in_sizes; (void)n_in; (void)out_size; (void)ws_size;

    const float* x      = (const float*)d_in[0];
    const float* wq_s   = (const float*)d_in[1];
    const float* wk_s   = (const float*)d_in[2];
    const float* wv_r   = (const float*)d_in[3];
    const float* wq_r   = (const float*)d_in[4];
    const float* wk_ret = (const float*)d_in[5];
    const float* w_out  = (const float*)d_in[6];
    float* out = (float*)d_out;

    const int MT = B_SZ * N_SEQ;      // 4096 rows
    const float scale = 0.125f;       // DH^-0.5

    char* ws = (char*)d_ws;
    size_t off = 0;
    auto alloc = [&](size_t bytes) -> void* {
        void* p = ws + off;
        off = (off + bytes + 255) & ~(size_t)255;
        return p;
    };

    __bf16* x_bf   = (__bf16*)alloc((size_t)MT * DMODEL * 2);
    __bf16* wqs_t  = (__bf16*)alloc((size_t)DMODEL * DMODEL * 2);
    __bf16* wks_t  = (__bf16*)alloc((size_t)DMODEL * DMODEL * 2);
    __bf16* wqr_t  = (__bf16*)alloc((size_t)DMODEL * DMODEL * 2);
    __bf16* wvr_t  = (__bf16*)alloc((size_t)DMODEL * (R_H * DIMH) * 2);
    __bf16* wout_t = (__bf16*)alloc((size_t)DMODEL * DMODEL * 2);
    __bf16* sqh    = (__bf16*)alloc((size_t)MT * DMODEL * 2);
    __bf16* skh    = (__bf16*)alloc((size_t)MT * DMODEL * 2);
    float*  rqf    = (float*) alloc((size_t)MT * DMODEL * 4);
    __bf16* rvh    = (__bf16*)alloc((size_t)MT * (R_H * DIMH) * 2);
    __bf16* rvt    = (__bf16*)alloc((size_t)MT * (R_H * DIMH) * 2);
    __bf16* osd    = (__bf16*)alloc((size_t)MT * DMODEL * 2);

    // --- precision conversions / weight transposes ---
    cvt_f32_bf16_kernel<<<(MT * DMODEL + 255) / 256, 256, 0, stream>>>(
        x, x_bf, MT * DMODEL);
    transpose_cvt_kernel<<<(DMODEL * DMODEL + 255) / 256, 256, 0, stream>>>(
        wq_s, wqs_t, DMODEL, DMODEL);
    transpose_cvt_kernel<<<(DMODEL * DMODEL + 255) / 256, 256, 0, stream>>>(
        wk_s, wks_t, DMODEL, DMODEL);
    transpose_cvt_kernel<<<(DMODEL * DMODEL + 255) / 256, 256, 0, stream>>>(
        wq_r, wqr_t, DMODEL, DMODEL);
    transpose_cvt_kernel<<<(DMODEL * R_H * DIMH + 255) / 256, 256, 0, stream>>>(
        wv_r, wvr_t, DMODEL, R_H * DIMH);
    transpose_cvt_kernel<<<(DMODEL * DMODEL + 255) / 256, 256, 0, stream>>>(
        w_out, wout_t, DMODEL, DMODEL);

    // --- projections (WMMA GEMMs) ---
    gemm_bf16_kernel<<<dim3(MT / 64, DMODEL / 64), 128, 0, stream>>>(
        x_bf, wqs_t, nullptr, sqh, MT, DMODEL, DMODEL, scale);        // sq (scaled)
    gemm_bf16_kernel<<<dim3(MT / 64, DMODEL / 64), 128, 0, stream>>>(
        x_bf, wks_t, nullptr, skh, MT, DMODEL, DMODEL, 1.0f);         // sk
    gemm_bf16_kernel<<<dim3(MT / 64, DMODEL / 64), 128, 0, stream>>>(
        x_bf, wqr_t, rqf, nullptr, MT, DMODEL, DMODEL, scale);        // rq (f32, scaled)
    gemm_bf16_kernel<<<dim3(MT / 64, (R_H * DIMH) / 64), 128, 0, stream>>>(
        x_bf, wvr_t, nullptr, rvh, MT, R_H * DIMH, DMODEL, 1.0f);     // rv

    rv_transpose_kernel<<<(B_SZ * R_H * DIMH * N_SEQ + 255) / 256, 256, 0,
                          stream>>>(rvh, rvt);

    // --- fused flash attention + retrieval composition ---
    attn_kernel<<<dim3(N_SEQ / 64, B_SZ * S_H), 128, 0, stream>>>(
        sqh, skh, rvt, rqf, wk_ret, osd);

    // --- final output projection ---
    gemm_bf16_kernel<<<dim3(MT / 64, DMODEL / 64), 128, 0, stream>>>(
        osd, wout_t, out, nullptr, MT, DMODEL, DMODEL, 1.0f);
}
